// HungrianHead_35673998360844
// MI455X (gfx1250) — compile-verified, matrix operand order checked by reading, not run on hardware
//
#include <hip/hip_runtime.h>
#include <hip/hip_bf16.h>

// ---------------- problem constants ----------------
#define NB   256          // batch
#define NP   192          // patches per image
#define NDK  768          // DV == DT (inner dim)
#define NE   256          // embed dim
#define NA   8            // attributes per image
#define RK   4            // top-k assignment rounds
#define HM   NP           // hungarian columns
#define HN   NA           // hungarian rows

#define LDA  776          // padded LDS row stride in ushorts (388 dwords % 64 banks = 4)
#define LDA4 (LDA / 8)    // 97 16-byte chunks per padded row

typedef __attribute__((ext_vector_type(16))) __bf16 v16bf;
typedef __attribute__((ext_vector_type(8)))  float  v8f;

union FragBF { v16bf v; uint4 q[2]; };

// Raw clang vector type matching the async-LDS builtin's expected parameter
// type: 'int __attribute__((vector_size(16)))' with explicit address spaces.
typedef int v4i_raw __attribute__((vector_size(4 * sizeof(int))));
typedef __attribute__((address_space(1))) v4i_raw gv4i;   // global
typedef __attribute__((address_space(3))) v4i_raw lv4i;   // LDS

#if __has_builtin(__builtin_amdgcn_global_load_async_to_lds_b128)
#define HAVE_ASYNC_LDS 1
#else
#define HAVE_ASYNC_LDS 0
#endif

__device__ __forceinline__ void wait_async_zero() {
#if __has_builtin(__builtin_amdgcn_s_wait_asynccnt)
    __builtin_amdgcn_s_wait_asynccnt(0);
#else
    asm volatile("s_wait_asynccnt 0" ::: "memory");
#endif
}

// round-to-nearest-even f32 -> bf16 (bit pattern as ushort)
__device__ __forceinline__ unsigned short f2bf(float f) {
    unsigned int u = __float_as_uint(f);
    unsigned int rnd = 0x7FFFu + ((u >> 16) & 1u);
    return (unsigned short)((u + rnd) >> 16);
}

// ---------------- conversion kernels ----------------
// W [768,256] f32 -> WT [256,768] bf16
__global__ void cvt_transpose_w(const float* __restrict__ W, unsigned short* __restrict__ WT) {
    int idx = blockIdx.x * blockDim.x + threadIdx.x;   // over 768*256
    if (idx >= NDK * NE) return;
    int k = idx / NE, n = idx % NE;
    WT[(size_t)n * NDK + k] = f2bf(W[idx]);
}

// generic f32 -> bf16
__global__ void cvt_bf16(const float* __restrict__ x, unsigned short* __restrict__ y, int n) {
    int i = blockIdx.x * blockDim.x + threadIdx.x;
    if (i < n) y[i] = f2bf(x[i]);
}

// visual_feature [B,193,768] -> Xv [B,768] (row 0) and Xp [B*192,768] (rows 1..192), bf16
__global__ void split_visual(const float* __restrict__ vf,
                             unsigned short* __restrict__ Xv,
                             unsigned short* __restrict__ Xp) {
    size_t idx = (size_t)blockIdx.x * blockDim.x + threadIdx.x;  // over B*193*768
    if (idx >= (size_t)NB * 193 * NDK) return;
    int d = (int)(idx % NDK);
    size_t rp = idx / NDK;
    int row = (int)(rp % 193);
    int b   = (int)(rp / 193);
    unsigned short h = f2bf(vf[idx]);
    if (row == 0) Xv[(size_t)b * NDK + d] = h;
    else          Xp[((size_t)b * NP + (row - 1)) * NDK + d] = h;
}

// ---------------- WMMA bf16 GEMM:  Y[M,256] = A[M,768] * WT^T + bias ----------------
// One block per 16-row strip of A. The strip (16x768 bf16, 24 KB) is staged into
// LDS once with async global->LDS b128 copies (ASYNCcnt), then 8 waves each
// compute two 16x16 n-tiles, reading A fragments from LDS (ds_load_b128, rows
// padded to kill bank conflicts) and B fragments from the L2-resident transposed
// weights. 24 k-steps of v_wmma_f32_16x16x32_bf16 per tile.
__global__ void gemm_wmma_bf16(const unsigned short* __restrict__ Abf,
                               const unsigned short* __restrict__ BTbf,
                               const float* __restrict__ bias,
                               float* __restrict__ Y, int M) {
    __shared__ unsigned short sA[16 * LDA];

    int tid  = threadIdx.x;
    int wave = tid >> 5;
    int lane = tid & 31;
    int half = (lane >> 4) & 1;
    int r    = lane & 15;

    int tile_m = blockIdx.x;              // one 16-row strip per block
    // ---- stage A strip into LDS (96 16B chunks per row, padded to 97 in LDS) ----
    {
        const uint4* gsrc = (const uint4*)(Abf + (size_t)tile_m * 16 * NDK);
#if HAVE_ASYNC_LDS
        gv4i* gp = (gv4i*)(unsigned long long)gsrc;
        lv4i* lp = (lv4i*)(unsigned int)(unsigned long long)(void*)sA;
        for (int t = tid; t < 16 * 96; t += 256) {
            int row = t / 96, c = t % 96;
            __builtin_amdgcn_global_load_async_to_lds_b128(gp + t, lp + row * LDA4 + c, 0, 0);
        }
        wait_async_zero();
#else
        for (int t = tid; t < 16 * 96; t += 256) {
            int row = t / 96, c = t % 96;
            ((uint4*)sA)[row * LDA4 + c] = gsrc[t];
        }
#endif
    }
    __syncthreads();

    // ---- each wave owns n-tiles (2*wave) and (2*wave+1): 32 output columns ----
    int n0 = wave * 32;
    const unsigned short* bp0 = BTbf + (size_t)(n0 + r)      * NDK + half * 8;
    const unsigned short* bp1 = BTbf + (size_t)(n0 + 16 + r) * NDK + half * 8;
    const unsigned short* apL = sA + r * LDA + half * 8;   // LDS fragment base

    v8f acc0 = {};
    v8f acc1 = {};
    for (int k0 = 0; k0 < NDK; k0 += 32) {
        FragBF a, b0, b1;
        // A layout (16-bit, 16x32): lanes 0-15 -> K = k0+0..7 and k0+16..23
        //                           lanes16-31 -> K = k0+8..15 and k0+24..31
        a.q[0]  = *(const uint4*)(apL + k0);        // ds_load_b128
        a.q[1]  = *(const uint4*)(apL + k0 + 16);
        b0.q[0] = *(const uint4*)(bp0 + k0);
        b0.q[1] = *(const uint4*)(bp0 + k0 + 16);
        b1.q[0] = *(const uint4*)(bp1 + k0);
        b1.q[1] = *(const uint4*)(bp1 + k0 + 16);
        if (k0 + 32 < NDK) {
            __builtin_prefetch(bp0 + k0 + 32, 0, 0);   // global_prefetch_b8
            __builtin_prefetch(bp1 + k0 + 32, 0, 0);
        }
        acc0 = __builtin_amdgcn_wmma_f32_16x16x32_bf16(
            false, a.v, false, b0.v, (short)0, acc0, false, false);
        acc1 = __builtin_amdgcn_wmma_f32_16x16x32_bf16(
            false, a.v, false, b1.v, (short)0, acc1, false, false);
    }

    float bv0 = bias[n0 + r];
    float bv1 = bias[n0 + 16 + r];
    // D layout: VGPR v -> row M = v + half*8, col N = r
    #pragma unroll
    for (int v = 0; v < 8; v++) {
        int m = tile_m * 16 + half * 8 + v;
        Y[(size_t)m * NE + n0 + r]      = acc0[v] + bv0;
        Y[(size_t)m * NE + n0 + 16 + r] = acc1[v] + bv1;
    }
    (void)M;
}

// ---------------- row-wise L2 normalize (one wave per 256-float row) ----------------
__global__ void l2norm_rows(float* __restrict__ X, int M) {
    int row  = blockIdx.x * 8 + (threadIdx.x >> 5);
    int lane = threadIdx.x & 31;
    if (row >= M) return;
    float* p = X + (size_t)row * NE;
    float s = 0.f;
    for (int e = lane; e < NE; e += 32) { float x = p[e]; s += x * x; }
    for (int off = 16; off > 0; off >>= 1) s += __shfl_xor(s, off, 32);
    float inv = 1.0f / fmaxf(sqrtf(s), 1e-12f);
    for (int e = lane; e < NE; e += 32) p[e] *= inv;
}

// ---------------- sim[b,a,p] = att[b,a,:] . pat[b,p,:] ----------------
__global__ void sim_kernel(const float* __restrict__ att, const float* __restrict__ pat,
                           float* __restrict__ sim) {
    __shared__ float sAtt[NA * NE];
    int b = blockIdx.x, tid = threadIdx.x;
    for (int t = tid; t < NA * NE; t += 256) sAtt[t] = att[(size_t)b * NA * NE + t];
    __syncthreads();
    if (tid < NP) {
        const float* pr = pat + ((size_t)b * NP + tid) * NE;
        float acc[NA];
        #pragma unroll
        for (int a = 0; a < NA; a++) acc[a] = 0.f;
        for (int e = 0; e < NE; e++) {
            float x = pr[e];
            #pragma unroll
            for (int a = 0; a < NA; a++) acc[a] += x * sAtt[a * NE + e];
        }
        #pragma unroll
        for (int a = 0; a < NA; a++) sim[((size_t)b * NA + a) * NP + tid] = acc[a];
    }
}

// ---------------- Hungarian (Jonker-Volgenant), one wave32 per batch ----------------
__global__ void hungarian_kernel(const float* __restrict__ sim, float* __restrict__ out_ls) {
    __shared__ float s_sim[HN * HM];
    __shared__ float s_v[HM + 1];
    __shared__ float s_minv[HM + 1];
    __shared__ int   s_way[HM + 1];
    __shared__ int   s_p[HM + 1];
    __shared__ int   s_used[HM + 1];
    __shared__ float s_u[HN + 1];

    const float INF = 1e30f;
    int b = blockIdx.x;
    int lane = threadIdx.x;   // 0..31

    for (int t = lane; t < HN * HM; t += 32) s_sim[t] = sim[(size_t)b * HN * HM + t];
    __syncthreads();

    float total = 0.f;
    for (int round = 0; round < RK; round++) {
        // fresh potentials / matching each round
        for (int t = lane; t <= HM; t += 32) { s_v[t] = 0.f; s_p[t] = 0; }
        if (lane <= HN) s_u[lane] = 0.f;
        __syncthreads();

        for (int i = 1; i <= HN; i++) {
            if (lane == 0) s_p[0] = i;
            for (int t = lane; t <= HM; t += 32) { s_minv[t] = INF; s_used[t] = 0; }
            __syncthreads();

            int j0 = 0;
            while (true) {
                if (lane == 0) s_used[j0] = 1;
                __syncthreads();
                int   i0  = s_p[j0];
                float ui0 = s_u[i0];

                // update minv over free cols + find (min, argmin) with numpy tie-break
                float bestv = INF;
                int   bestj = 0x7FFFFFFF;
                for (int t = lane; t < HM; t += 32) {
                    int j = t + 1;
                    if (!s_used[j]) {
                        float cur = -s_sim[(i0 - 1) * HM + t] - ui0 - s_v[j];
                        if (cur < s_minv[j]) { s_minv[j] = cur; s_way[j] = j0; }
                        float mv = s_minv[j];
                        if (mv < bestv) { bestv = mv; bestj = j; }
                    }
                }
                for (int off = 16; off > 0; off >>= 1) {
                    float ov = __shfl_xor(bestv, off, 32);
                    int   oj = __shfl_xor(bestj, off, 32);
                    if (ov < bestv || (ov == bestv && oj < bestj)) { bestv = ov; bestj = oj; }
                }
                float delta = bestv;
                int   j1    = bestj;
                __syncthreads();

                // u[p[used]] += delta; v[used] -= delta; minv[free] -= delta
                for (int t = lane; t <= HM; t += 32) {
                    if (s_used[t]) { s_u[s_p[t]] += delta; s_v[t] -= delta; }
                    else if (t >= 1) s_minv[t] -= delta;
                }
                __syncthreads();

                j0 = j1;
                if (s_p[j0] == 0) break;
            }
            // augmenting path (serial, lane 0)
            if (lane == 0) {
                int jj = j0;
                while (jj != 0) { int jn = s_way[jj]; s_p[jj] = s_p[jn]; jj = jn; }
            }
            __syncthreads();
        }

        // accumulate matched sims, then mask them to -100
        float lsum = 0.f;
        for (int t = lane; t < HM; t += 32) {
            int pi = s_p[t + 1];
            if (pi != 0) lsum += s_sim[(pi - 1) * HM + t];
        }
        for (int off = 16; off > 0; off >>= 1) lsum += __shfl_xor(lsum, off, 32);
        __syncthreads();
        for (int t = lane; t < HM; t += 32) {
            int pi = s_p[t + 1];
            if (pi != 0) s_sim[(pi - 1) * HM + t] = -100.f;
        }
        __syncthreads();
        total += lsum / (float)HN;
    }
    if (lane == 0) out_ls[b] = total / (float)RK;
}

// ---------------- launcher ----------------
extern "C" void kernel_launch(void* const* d_in, const int* in_sizes, int n_in,
                              void* d_out, int out_size, void* d_ws, size_t ws_size,
                              hipStream_t stream) {
    const float* vf  = (const float*)d_in[0];   // [B,193,768]
    const float* tf  = (const float*)d_in[1];   // [B,768]
    const float* af  = (const float*)d_in[2];   // [B*8,768]
    const float* Wv  = (const float*)d_in[4];
    const float* bv  = (const float*)d_in[5];
    const float* Wt  = (const float*)d_in[6];
    const float* bt  = (const float*)d_in[7];
    const float* Wp  = (const float*)d_in[8];
    const float* bp  = (const float*)d_in[9];
    const float* Wa  = (const float*)d_in[10];
    const float* ba  = (const float*)d_in[11];

    float* out_vis = (float*)d_out;                       // [B,E]
    float* out_txt = (float*)d_out + (size_t)NB * NE;     // [B,E]
    float* out_ls  = (float*)d_out + (size_t)2 * NB * NE; // [B]

    // workspace layout (bytes)
    char* ws = (char*)d_ws;
    const size_t WT_BYTES = (size_t)NE * NDK * 2;           // 393216
    unsigned short* wsWvT = (unsigned short*)(ws + 0 * WT_BYTES);
    unsigned short* wsWtT = (unsigned short*)(ws + 1 * WT_BYTES);
    unsigned short* wsWpT = (unsigned short*)(ws + 2 * WT_BYTES);
    unsigned short* wsWaT = (unsigned short*)(ws + 3 * WT_BYTES);
    size_t off = 4 * WT_BYTES;
    unsigned short* wsXv = (unsigned short*)(ws + off); off += (size_t)NB * NDK * 2;
    unsigned short* wsXt = (unsigned short*)(ws + off); off += (size_t)NB * NDK * 2;
    unsigned short* wsXa = (unsigned short*)(ws + off); off += (size_t)NB * NA * NDK * 2;
    unsigned short* wsXp = (unsigned short*)(ws + off); off += (size_t)NB * NP * NDK * 2;
    float* wsPat = (float*)(ws + off); off += (size_t)NB * NP * NE * 4;
    float* wsAtt = (float*)(ws + off); off += (size_t)NB * NA * NE * 4;
    float* wsSim = (float*)(ws + off); off += (size_t)NB * NA * NP * 4;

    // 1) weights -> transposed bf16
    {
        int n = NDK * NE;
        int blk = (n + 255) / 256;
        cvt_transpose_w<<<blk, 256, 0, stream>>>(Wv, wsWvT);
        cvt_transpose_w<<<blk, 256, 0, stream>>>(Wt, wsWtT);
        cvt_transpose_w<<<blk, 256, 0, stream>>>(Wp, wsWpT);
        cvt_transpose_w<<<blk, 256, 0, stream>>>(Wa, wsWaT);
    }
    // 2) activations -> bf16
    {
        size_t n = (size_t)NB * 193 * NDK;
        split_visual<<<(unsigned)((n + 255) / 256), 256, 0, stream>>>(vf, wsXv, wsXp);
        int nt = NB * NDK;
        cvt_bf16<<<(nt + 255) / 256, 256, 0, stream>>>(tf, wsXt, nt);
        int na = NB * NA * NDK;
        cvt_bf16<<<(na + 255) / 256, 256, 0, stream>>>(af, wsXa, na);
    }
    // 3) WMMA GEMMs: one block per 16-row strip (block covers all of N=256)
    {
        gemm_wmma_bf16<<<NB / 16,        256, 0, stream>>>(wsXv, wsWvT, bv, out_vis, NB);
        gemm_wmma_bf16<<<NB / 16,        256, 0, stream>>>(wsXt, wsWtT, bt, out_txt, NB);
        gemm_wmma_bf16<<<NB * NP / 16,   256, 0, stream>>>(wsXp, wsWpT, bp, wsPat, NB * NP);
        gemm_wmma_bf16<<<NB * NA / 16,   256, 0, stream>>>(wsXa, wsWaT, ba, wsAtt, NB * NA);
    }
    // 4) L2 normalize patch + attribute embeds
    {
        int Mp = NB * NP;
        l2norm_rows<<<(Mp + 7) / 8, 256, 0, stream>>>(wsPat, Mp);
        int Ma = NB * NA;
        l2norm_rows<<<(Ma + 7) / 8, 256, 0, stream>>>(wsAtt, Ma);
    }
    // 5) similarity
    sim_kernel<<<NB, 256, 0, stream>>>(wsAtt, wsPat, wsSim);
    // 6) hungarian top-k assignment -> local_similarity
    hungarian_kernel<<<NB, 32, 0, stream>>>(wsSim, out_ls);

    (void)in_sizes; (void)n_in; (void)out_size; (void)ws_size;
}
